// CausalSelfAttention_61512521614032
// MI455X (gfx1250) — compile-verified, matrix operand order checked by reading, not run on hardware
//
#include <hip/hip_runtime.h>

// Problem constants (reference: B=4, T=2048, C=2048, H=16 -> D=128)
#define BATCH 4
#define SEQ   2048
#define CH    2048
#define N_HEAD 16
#define HDIM  (CH / N_HEAD)   // 128

typedef __bf16 bf16;
typedef __attribute__((ext_vector_type(16))) __bf16 v16bf;
typedef __attribute__((ext_vector_type(8)))  __bf16 v8bf;
typedef __attribute__((ext_vector_type(8)))  float  v8f;
typedef __attribute__((ext_vector_type(4)))  int    v4i;

static __device__ __forceinline__ v8f wmma_bf16(v16bf a, v16bf b, v8f c) {
  // D(f32 16x16) = A(bf16 16x32) * B(bf16 32x16) + C
  return __builtin_amdgcn_wmma_f32_16x16x32_bf16(false, a, false, b, (short)0, c,
                                                 false, false);
}

static __device__ __forceinline__ v8f zero8() {
  v8f z = {0.f, 0.f, 0.f, 0.f, 0.f, 0.f, 0.f, 0.f};
  return z;
}

// Load one lane's 16 bf16 elements of a 16x32 A-fragment (or 32x16 B-fragment,
// column-contiguous). Per the CDNA5 16-bit layout, lane l holds contiguous
// K = koff..koff+7 (VGPR0-3) and K = koff+16..koff+23 (VGPR4-7), koff = (l>=16)?8:0.
// Caller passes p already offset by koff; p must be 16B aligned.
static __device__ __forceinline__ v16bf load_frag(const bf16* p) {
  v8bf lo = *reinterpret_cast<const v8bf*>(p);
  v8bf hi = *reinterpret_cast<const v8bf*>(p + 16);
  v16bf r;
#pragma unroll
  for (int e = 0; e < 8; ++e) { r[e] = lo[e]; r[e + 8] = hi[e]; }
  return r;
}

static __device__ __forceinline__ float half_reduce_max(float v) {
#pragma unroll
  for (int m = 1; m < 16; m <<= 1) v = fmaxf(v, __shfl_xor(v, m, 32));
  return v;
}
static __device__ __forceinline__ float half_reduce_sum(float v) {
#pragma unroll
  for (int m = 1; m < 16; m <<= 1) v += __shfl_xor(v, m, 32);
  return v;
}

// ---------------------------------------------------------------------------
// Async global -> LDS copy (CDNA5 ASYNCcnt path), with safe fallback.
// The builtin's first parameter is 'v4i __device__ *' (16-byte int vector in
// the global address space) per the compiler diagnostic; LDS dest is AS(3).
// ---------------------------------------------------------------------------
#if defined(__has_builtin)
#if __has_builtin(__builtin_amdgcn_global_load_async_to_lds_b128)
#define HAVE_ASYNC_LDS 1
#endif
#endif

typedef __attribute__((address_space(1))) v4i gas_v4i;
typedef __attribute__((address_space(3))) v4i las_v4i;

static __device__ __forceinline__ void async_copy16(const bf16* g, bf16* l) {
#ifdef HAVE_ASYNC_LDS
  __builtin_amdgcn_global_load_async_to_lds_b128(
      (gas_v4i*)(void*)g, (las_v4i*)(void*)l, 0, 0);
#else
  *reinterpret_cast<v8bf*>(l) = *reinterpret_cast<const v8bf*>(g);
#endif
}

static __device__ __forceinline__ void async_join() {
#ifdef HAVE_ASYNC_LDS
#if __has_builtin(__builtin_amdgcn_s_wait_asynccnt)
  __builtin_amdgcn_s_wait_asynccnt(0);
#else
  asm volatile("s_wait_asynccnt 0x0" ::: "memory");
#endif
#endif
}

// ---------------------------------------------------------------------------
// 1) fp32 -> bf16 cast
// ---------------------------------------------------------------------------
__global__ __launch_bounds__(256) void cast_f32_to_bf16(const float* __restrict__ in,
                                                        bf16* __restrict__ out,
                                                        size_t n) {
  size_t i = (size_t)blockIdx.x * blockDim.x + threadIdx.x;
  if (i < n) out[i] = (bf16)in[i];
}

// ---------------------------------------------------------------------------
// 2) W[K][N] fp32  ->  Wt[N][K] bf16  (LDS-tiled transpose, coalesced both sides)
// ---------------------------------------------------------------------------
__global__ __launch_bounds__(256) void transpose_cast_bf16(const float* __restrict__ W,
                                                           bf16* __restrict__ Wt,
                                                           int K, int N) {
  __shared__ float tile[32][33];
  const int kb = blockIdx.x * 32;
  const int nb = blockIdx.y * 32;
  const int tx = threadIdx.x & 31;
  const int ty = threadIdx.x >> 5;   // 0..7
#pragma unroll
  for (int i = ty; i < 32; i += 8)
    tile[i][tx] = W[(size_t)(kb + i) * N + nb + tx];
  __syncthreads();
#pragma unroll
  for (int i = ty; i < 32; i += 8)
    Wt[(size_t)(nb + i) * K + kb + tx] = (bf16)tile[tx][i];
}

// ---------------------------------------------------------------------------
// 3) C[M][N] (fp32) = A[M][K] (bf16, row-major) * Bt[N][K]^T (bf16)
//    Register-blocked: each wave owns a 32x64 tile (2x4 grid of 16x16 WMMA
//    accumulators -> 8 WMMAs per K=32 step from only 6 fragment loads).
//    Next K-step fragments are loaded before the current step's WMMAs
//    (software pipeline). 8 waves/block -> 64 x 256 macro-tile.
// ---------------------------------------------------------------------------
__global__ __launch_bounds__(256) void gemm_bf16_nt(const bf16* __restrict__ A,
                                                    const bf16* __restrict__ Bt,
                                                    float* __restrict__ C,
                                                    int M, int N, int K) {
  const int lane = threadIdx.x & 31;
  const int wave = threadIdx.x >> 5;
  const int tile_m = blockIdx.x * 64 + (wave >> 2) * 32;   // wave M tile: 32
  const int tile_n = blockIdx.y * 256 + (wave & 3) * 64;   // wave N tile: 64
  const int l16 = lane & 15;
  const int koff = (lane >= 16) ? 8 : 0;

  const bf16* ap0 = A + (size_t)(tile_m + l16) * K + koff;
  const bf16* ap1 = ap0 + (size_t)16 * K;
  const bf16* bp0 = Bt + (size_t)(tile_n + l16) * K + koff;
  const bf16* bp1 = bp0 + (size_t)16 * K;
  const bf16* bp2 = bp0 + (size_t)32 * K;
  const bf16* bp3 = bp0 + (size_t)48 * K;

  v8f acc[2][4];
#pragma unroll
  for (int i = 0; i < 2; ++i)
#pragma unroll
    for (int j = 0; j < 4; ++j) acc[i][j] = zero8();

  // Prologue: load K-step 0 fragments.
  v16bf a0 = load_frag(ap0), a1 = load_frag(ap1);
  v16bf b0 = load_frag(bp0), b1 = load_frag(bp1);
  v16bf b2 = load_frag(bp2), b3 = load_frag(bp3);

  for (int kb = 0; kb < K; kb += 32) {
    // Issue next K-step loads before consuming the current fragments.
    v16bf na0 = a0, na1 = a1, nb0 = b0, nb1 = b1, nb2 = b2, nb3 = b3;
    const int kn = kb + 32;
    if (kn < K) {
      na0 = load_frag(ap0 + kn);
      na1 = load_frag(ap1 + kn);
      nb0 = load_frag(bp0 + kn);
      nb1 = load_frag(bp1 + kn);
      nb2 = load_frag(bp2 + kn);
      nb3 = load_frag(bp3 + kn);
      __builtin_prefetch(ap0 + kn + 32, 0, 3);   // -> global_prefetch_b8
      __builtin_prefetch(bp0 + kn + 32, 0, 3);
    }

    acc[0][0] = wmma_bf16(a0, b0, acc[0][0]);
    acc[0][1] = wmma_bf16(a0, b1, acc[0][1]);
    acc[0][2] = wmma_bf16(a0, b2, acc[0][2]);
    acc[0][3] = wmma_bf16(a0, b3, acc[0][3]);
    acc[1][0] = wmma_bf16(a1, b0, acc[1][0]);
    acc[1][1] = wmma_bf16(a1, b1, acc[1][1]);
    acc[1][2] = wmma_bf16(a1, b2, acc[1][2]);
    acc[1][3] = wmma_bf16(a1, b3, acc[1][3]);

    a0 = na0; a1 = na1; b0 = nb0; b1 = nb1; b2 = nb2; b3 = nb3;
  }

  // C/D layout: VGPR r holds M = base + r (+8 for lanes 16-31), N = lane&15.
  const int hh8 = (lane >= 16) ? 8 : 0;
#pragma unroll
  for (int i = 0; i < 2; ++i) {
    const int mb = tile_m + 16 * i + hh8;
#pragma unroll
    for (int j = 0; j < 4; ++j) {
      float* cp = C + (size_t)mb * N + tile_n + 16 * j + l16;
#pragma unroll
      for (int r = 0; r < 8; ++r) cp[(size_t)r * N] = acc[i][j][r];
    }
  }
}

// ---------------------------------------------------------------------------
// 4) RoPE on q,k + head repack. qkv fp32 [B,T,3C] ->
//    q,k bf16 [B,H,T,D] (roped), vT bf16 [B,H,D,T] (transposed for P@V).
// ---------------------------------------------------------------------------
__global__ __launch_bounds__(256) void rope_pack(const float* __restrict__ qkv,
                                                 bf16* __restrict__ Q,
                                                 bf16* __restrict__ Kmat,
                                                 bf16* __restrict__ Vt) {
  const size_t total = (size_t)BATCH * N_HEAD * SEQ * (HDIM / 2);
  size_t idx = (size_t)blockIdx.x * blockDim.x + threadIdx.x;
  if (idx >= total) return;
  const int i = (int)(idx % (HDIM / 2));
  size_t r = idx / (HDIM / 2);
  const int t = (int)(r % SEQ); r /= SEQ;
  const int h = (int)(r % N_HEAD);
  const int b = (int)(r / N_HEAD);

  const float* base = qkv + ((size_t)b * SEQ + t) * (3 * CH);
  const int col = h * HDIM + 2 * i;
  const float q0 = base[col],          q1 = base[col + 1];
  const float k0 = base[CH + col],     k1 = base[CH + col + 1];
  const float v0 = base[2 * CH + col], v1 = base[2 * CH + col + 1];

  // theta = base^{-2i/D}; angle = t * theta   (interleaved-pair RoPE)
  const float inv_freq = __expf(-((float)(2 * i) / (float)HDIM) * __logf(10000.0f));
  const float ang = (float)t * inv_freq;
  float s, c;
  __sincosf(ang, &s, &c);

  const size_t qo = (((size_t)b * N_HEAD + h) * SEQ + t) * HDIM + 2 * i;
  Q[qo]     = (bf16)(q0 * c - q1 * s);
  Q[qo + 1] = (bf16)(q0 * s + q1 * c);
  Kmat[qo]     = (bf16)(k0 * c - k1 * s);
  Kmat[qo + 1] = (bf16)(k0 * s + k1 * c);

  const size_t vo = (((size_t)b * N_HEAD + h) * HDIM + 2 * i) * SEQ + t;
  Vt[vo]       = (bf16)v0;
  Vt[vo + SEQ] = (bf16)v1;
}

// ---------------------------------------------------------------------------
// 5) Flash attention, cooperative + double-buffered.
//    Block = 4 waves = 64 query rows. Per 32-key iteration the block stages
//    the K tile (32x128) and V tile (128x32, from vT) into LDS with async
//    global->LDS copies (ASYNCcnt), double-buffered so the next tile's DMA
//    overlaps the current tile's WMMA + softmax. Each wave owns 16 q rows:
//    two 16x16 score accumulators -> K=32 A-fragment for P@V (staged via LDS).
// ---------------------------------------------------------------------------
#define FA_WAVES 4

__global__ __launch_bounds__(32 * FA_WAVES) void flash_attn_wmma(
    const bf16* __restrict__ Q, const bf16* __restrict__ Kmat,
    const bf16* __restrict__ Vt, bf16* __restrict__ Y, float scale) {
  __shared__ __align__(16) bf16 Kt[2][32][HDIM];      // 2 x 8 KB
  __shared__ __align__(16) bf16 Vts[2][HDIM][32];     // 2 x 8 KB
  __shared__ __align__(16) bf16 Pl[FA_WAVES][16][32]; // 4 KB

  const int tid = threadIdx.x;
  const int lane = tid & 31;
  const int wave = tid >> 5;
  const int l16 = lane & 15;
  const int hh8 = (lane >= 16) ? 8 : 0;     // C-layout row offset == K packing offset
  const int qbase = blockIdx.x * (16 * FA_WAVES);
  const int qw = qbase + 16 * wave;         // this wave's 16 query rows
  const int bh = blockIdx.y;

  const bf16* qh = Q    + (size_t)bh * SEQ * HDIM;
  const bf16* kh = Kmat + (size_t)bh * SEQ * HDIM;
  const bf16* vh = Vt   + (size_t)bh * HDIM * SEQ;

  // Query tile 16 x 128 kept in registers as 4 A-fragments (K=32 each).
  v16bf qa[4];
#pragma unroll
  for (int c = 0; c < 4; ++c)
    qa[c] = load_frag(qh + (size_t)(qw + l16) * HDIM + c * 32 + hh8);

  v8f O[8];
#pragma unroll
  for (int c = 0; c < 8; ++c) O[c] = zero8();
  float rm[8], rl[8];
#pragma unroll
  for (int r = 0; r < 8; ++r) { rm[r] = -3.0e38f; rl[r] = 0.0f; }

  // Cooperative staging of one 32-key block (K tile + V tile) into buffer buf.
  // 1024 16-byte chunks / 128 threads = 8 async copies per thread.
  auto stage = [&](int j, int buf) {
#pragma unroll
    for (int i = 0; i < 4; ++i) {
      const int chunk = tid + 128 * i;          // 0..511
      const int row = chunk >> 4;               // key row 0..31
      const int c16 = chunk & 15;               // 16B chunk within 256B row
      async_copy16(kh + (size_t)(j + row) * HDIM + c16 * 8, &Kt[buf][row][c16 * 8]);
    }
#pragma unroll
    for (int i = 0; i < 4; ++i) {
      const int chunk = tid + 128 * i;          // 0..511
      const int d = chunk >> 2;                 // dim row 0..127
      const int c4 = chunk & 3;                 // 16B chunk within 64B row
      async_copy16(vh + (size_t)d * SEQ + j + c4 * 8, &Vts[buf][d][c4 * 8]);
    }
  };

  const int jend = qbase + 16 * FA_WAVES;       // block-level causal bound
  stage(0, 0);
  async_join();
  __syncthreads();

  int cur = 0;
  for (int j = 0; j < jend; j += 32) {
    // Kick off next tile's DMA into the other buffer before computing.
    if (j + 32 < jend) stage(j + 32, cur ^ 1);

    // S = q . k^T  (16 x 32 scores as two 16x16 accumulators), K from LDS.
    v8f S0 = zero8(), S1 = zero8();
#pragma unroll
    for (int c = 0; c < 4; ++c) {
      v16bf b0 = load_frag(&Kt[cur][l16][c * 32 + hh8]);
      v16bf b1 = load_frag(&Kt[cur][16 + l16][c * 32 + hh8]);
      S0 = wmma_bf16(qa[c], b0, S0);
      S1 = wmma_bf16(qa[c], b1, S1);
    }

    // Online softmax. Each half-wave holds one full row per C-VGPR, so row
    // reductions are 16-lane butterflies. Fully-masked sub-blocks (waves
    // below the diagonal) degrade to alpha=1, p=0: harmless no-ops.
    const int kc0 = j + l16, kc1 = kc0 + 16;
    float alpha[8];
#pragma unroll
    for (int r = 0; r < 8; ++r) {
      const int mq = qw + r + hh8;
      float s0 = S0[r] * scale; if (kc0 > mq) s0 = -3.0e38f;
      float s1 = S1[r] * scale; if (kc1 > mq) s1 = -3.0e38f;
      const float mx   = half_reduce_max(fmaxf(s0, s1));
      const float mnew = fmaxf(rm[r], mx);
      const float a    = __expf(rm[r] - mnew);
      const float p0   = __expf(s0 - mnew);
      const float p1   = __expf(s1 - mnew);
      const float rs   = half_reduce_sum(p0 + p1);
      rl[r] = rl[r] * a + rs;
      rm[r] = mnew;
      alpha[r] = a;
      Pl[wave][r + hh8][l16]      = (bf16)p0;   // per-wave row-major 16x32 P
      Pl[wave][r + hh8][l16 + 16] = (bf16)p1;
    }
#pragma unroll
    for (int c = 0; c < 8; ++c)
#pragma unroll
      for (int r = 0; r < 8; ++r) O[c][r] *= alpha[r];

    // LDS ops of one wave complete in order: P stores precede this reload.
    const v16bf pa = load_frag(&Pl[wave][l16][hh8]);

    // O += P @ V : B-fragment columns are vT rows (contiguous along keys).
#pragma unroll
    for (int c = 0; c < 8; ++c) {
      v16bf vb = load_frag(&Vts[cur][c * 16 + l16][hh8]);
      O[c] = wmma_bf16(pa, vb, O[c]);
    }

    async_join();        // this wave's DMA for buffer cur^1 is complete
    __syncthreads();     // everyone done reading cur / writing cur^1
    cur ^= 1;
  }

  // Normalize and write y in [B,T,C] bf16 for the output projection GEMM.
  const int b = bh / N_HEAD, h = bh % N_HEAD;
#pragma unroll
  for (int r = 0; r < 8; ++r) {
    const float inv = 1.0f / rl[r];
    bf16* yp = Y + ((size_t)b * SEQ + (qw + r + hh8)) * CH + h * HDIM + l16;
#pragma unroll
    for (int c = 0; c < 8; ++c) yp[c * 16] = (bf16)(O[c][r] * inv);
  }
}

// ---------------------------------------------------------------------------
// Host orchestration
// ---------------------------------------------------------------------------
extern "C" void kernel_launch(void* const* d_in, const int* in_sizes, int n_in,
                              void* d_out, int out_size, void* d_ws, size_t ws_size,
                              hipStream_t stream) {
  (void)in_sizes; (void)n_in; (void)out_size; (void)ws_size;
  const float* x     = (const float*)d_in[0];
  const float* Wqkv  = (const float*)d_in[1];
  const float* Wproj = (const float*)d_in[2];
  float* out = (float*)d_out;

  const int B = BATCH, T = SEQ, C = CH, H = N_HEAD, D = HDIM;
  const size_t BT = (size_t)B * T;

  // Workspace carve-out (~384 MB total)
  char* ws = (char*)d_ws;
  size_t off = 0;
  auto carve = [&](size_t bytes) -> char* {
    char* p = ws + off;
    off += (bytes + 255) & ~(size_t)255;
    return p;
  };
  bf16*  xbf    = (bf16*) carve(BT * C * sizeof(bf16));            // 32 MB
  bf16*  wqkvT  = (bf16*) carve((size_t)3 * C * C * sizeof(bf16)); // 24 MB
  bf16*  wprojT = (bf16*) carve((size_t)C * C * sizeof(bf16));     //  8 MB
  float* qkv    = (float*)carve(BT * 3 * C * sizeof(float));       // 192 MB
  bf16*  qbf    = (bf16*) carve(BT * C * sizeof(bf16));            // 32 MB
  bf16*  kbf    = (bf16*) carve(BT * C * sizeof(bf16));            // 32 MB
  bf16*  vtbf   = (bf16*) carve(BT * C * sizeof(bf16));            // 32 MB
  bf16*  ybf    = (bf16*) carve(BT * C * sizeof(bf16));            // 32 MB

  // 1) x -> bf16
  {
    const size_t n = BT * C;
    cast_f32_to_bf16<<<dim3((unsigned)((n + 255) / 256)), dim3(256), 0, stream>>>(x, xbf, n);
  }
  // 2) weight transposes (bf16, [N][K])
  transpose_cast_bf16<<<dim3(C / 32, (3 * C) / 32), dim3(256), 0, stream>>>(Wqkv, wqkvT, C, 3 * C);
  transpose_cast_bf16<<<dim3(C / 32, C / 32), dim3(256), 0, stream>>>(Wproj, wprojT, C, C);
  // 3) fused QKV projection: [8192,2048] x [2048,6144] -> fp32
  gemm_bf16_nt<<<dim3((unsigned)(BT / 64), (3 * C) / 256), dim3(256), 0, stream>>>(
      xbf, wqkvT, qkv, (int)BT, 3 * C, C);
  // 4) RoPE + head repack
  {
    const size_t n = (size_t)B * H * T * (D / 2);
    rope_pack<<<dim3((unsigned)((n + 255) / 256)), dim3(256), 0, stream>>>(qkv, qbf, kbf, vtbf);
  }
  // 5) causal flash attention (WMMA, async double-buffered K/V staging)
  flash_attn_wmma<<<dim3(T / (16 * FA_WAVES), B * H), dim3(32 * FA_WAVES), 0, stream>>>(
      qbf, kbf, vtbf, ybf, 1.0f / sqrtf((float)D));
  // 6) output projection -> fp32 d_out
  gemm_bf16_nt<<<dim3((unsigned)(BT / 64), C / 256), dim3(256), 0, stream>>>(
      ybf, wprojT, out, (int)BT, C, C);
}
